// TransMIL_24412594110494
// MI455X (gfx1250) — matrix operand order repaired and checked by reading, not compiled, not run on hardware
//
#include <hip/hip_runtime.h>

// ---------------------------------------------------------------- constants
#define DIMC      512
#define HEADS     8
#define DH        64
#define LM        256       // landmarks
#define N0        15000     // raw tokens
#define HW_       123       // ceil(sqrt(15000))
#define NFEAT     (HW_*HW_) // 15129
#define NTOK      (NFEAT+1) // 15130 (cls + feat)
#define PADR      230       // (256 - 15130%256)%256
#define NP        15360     // padded token count
#define LPER      60        // NP / LM
#define RESK      33
#define QKVW      1536

typedef __bf16 bf16x16 __attribute__((ext_vector_type(16)));
typedef float  f32x8   __attribute__((ext_vector_type(8)));

enum { FLAG_BIAS = 1, FLAG_RELU = 2, FLAG_AI = 4, FLAG_SCALE = 8, FLAG_ADD = 16 };

__device__ __forceinline__ unsigned short f2bf(float f) {
    unsigned int u = __float_as_uint(f);
    u += 0x7fffu + ((u >> 16) & 1u);      // round-to-nearest-even
    return (unsigned short)(u >> 16);
}
__device__ __forceinline__ unsigned int pk2(float a, float b) {
    return (unsigned int)f2bf(a) | ((unsigned int)f2bf(b) << 16);
}

// ---------------------------------------------------------------- generic WMMA GEMM
// C[M,N] = epilogue( A[M,K] @ B )   A row-major (lda), B element (k,n) at B[k*ldbk + n*ldbn]
// 128 threads (4 waves), 64x64 tile, K-step 32, bf16 inputs, f32 accumulate.
// Requires K % 32 == 0 and (ldbn==1 || ldbk==1) for the fast path (true for all call sites).
__global__ void __launch_bounds__(128)
gemm_wmma(const float* __restrict__ A, int lda, long long sA,
          const float* __restrict__ B, int ldbk, int ldbn, long long sB,
          float* __restrict__ C, int ldc, long long sC,
          const float* __restrict__ bias,
          int M, int N, int K, int flags, float alpha, int row_off)
{
    __shared__ unsigned short As[64 * 32];   // [m][k]
    __shared__ unsigned short Bs[64 * 32];   // [n][k]  (transposed)

    const int z = blockIdx.z;
    A += (long long)z * sA; B += (long long)z * sB; C += (long long)z * sC;

    const int m0 = blockIdx.y * 64, n0 = blockIdx.x * 64;
    const int tid = threadIdx.x, wave = tid >> 5, lane = tid & 31;
    const int r = lane & 15, half = lane >> 4;
    const bool fast = (m0 + 64 <= M) && (n0 + 64 <= N);

    f32x8 acc[4];
    for (int t = 0; t < 4; ++t)
        for (int e = 0; e < 8; ++e) acc[t][e] = 0.f;

    for (int k0 = 0; k0 < K; k0 += 32) {
        __syncthreads();
        if (fast) {
            // ---- A: row = tid>>1, 16 contiguous k per thread (global b128, lds b128)
            {
                int row = tid >> 1, seg = (tid & 1) * 16;
                const float* src = A + (size_t)(m0 + row) * lda + k0 + seg;
                const float4* s4 = (const float4*)src;
                float4 f0 = s4[0], f1 = s4[1], f2 = s4[2], f3 = s4[3];
                uint4 p0, p1;
                p0.x = pk2(f0.x, f0.y); p0.y = pk2(f0.z, f0.w);
                p0.z = pk2(f1.x, f1.y); p0.w = pk2(f1.z, f1.w);
                p1.x = pk2(f2.x, f2.y); p1.y = pk2(f2.z, f2.w);
                p1.z = pk2(f3.x, f3.y); p1.w = pk2(f3.z, f3.w);
                uint4* d = (uint4*)(As + row * 32 + seg);
                d[0] = p0; d[1] = p1;
                if (k0 + 32 < K) __builtin_prefetch(src + 32, 0, 0);   // next K tile
            }
            // ---- B
            if (ldbn == 1) {       // row-major weights: contiguous along n, scatter into Bs[n][k]
                int k = tid >> 2, nseg = (tid & 3) * 16;
                const float* src = B + (size_t)(k0 + k) * ldbk + n0 + nseg;
                const float4* s4 = (const float4*)src;
                float4 f0 = s4[0], f1 = s4[1], f2 = s4[2], f3 = s4[3];
                float v[16] = { f0.x,f0.y,f0.z,f0.w, f1.x,f1.y,f1.z,f1.w,
                                f2.x,f2.y,f2.z,f2.w, f3.x,f3.y,f3.z,f3.w };
#pragma unroll
                for (int j = 0; j < 16; ++j) Bs[(nseg + j) * 32 + k] = f2bf(v[j]);
                if (k0 + 32 < K) __builtin_prefetch(src + (size_t)32 * ldbk, 0, 0);
            } else {               // ldbk==1: contiguous along k, packed stores
                int n = tid >> 1, seg = (tid & 1) * 16;
                const float* src = B + (size_t)(k0 + seg) + (size_t)(n0 + n) * ldbn;
                const float4* s4 = (const float4*)src;
                float4 f0 = s4[0], f1 = s4[1], f2 = s4[2], f3 = s4[3];
                uint4 p0, p1;
                p0.x = pk2(f0.x, f0.y); p0.y = pk2(f0.z, f0.w);
                p0.z = pk2(f1.x, f1.y); p0.w = pk2(f1.z, f1.w);
                p1.x = pk2(f2.x, f2.y); p1.y = pk2(f2.z, f2.w);
                p1.z = pk2(f3.x, f3.y); p1.w = pk2(f3.z, f3.w);
                uint4* d = (uint4*)(Bs + n * 32 + seg);
                d[0] = p0; d[1] = p1;
            }
        } else {
            for (int i = 0; i < 16; ++i) {
                int idx = i * 128 + tid;              // 0..2047
                int row = idx >> 5, col = idx & 31;
                int gm = m0 + row, gk = k0 + col, gn = n0 + row;
                As[idx] = (gm < M && gk < K) ? f2bf(A[(size_t)gm * lda + gk]) : (unsigned short)0;
                Bs[idx] = (gn < N && gk < K) ? f2bf(B[(size_t)gk * ldbk + (size_t)gn * ldbn]) : (unsigned short)0;
            }
        }
        __syncthreads();

        union { bf16x16 v; unsigned short u[16]; } af;
#pragma unroll
        for (int i = 0; i < 16; ++i) {
            int kk = (half << 4) + i;   // A 16x32 bf16 lane layout
            af.u[i] = As[(wave * 16 + r) * 32 + kk];
        }
#pragma unroll
        for (int nt = 0; nt < 4; ++nt) {
            union { bf16x16 v; unsigned short u[16]; } bfr;
#pragma unroll
            for (int i = 0; i < 16; ++i) {
                int kk = (half << 4) + i;                        // B 32x16 bf16 lane layout
                bfr.u[i] = Bs[(nt * 16 + r) * 32 + kk];
            }
            acc[nt] = __builtin_amdgcn_wmma_f32_16x16x32_bf16(
                false, af.v, false, bfr.v, (short)0, acc[nt], false, false);
        }
    }

#pragma unroll
    for (int nt = 0; nt < 4; ++nt)
#pragma unroll
        for (int j = 0; j < 8; ++j) {
            int gm = m0 + wave * 16 + j + (half << 3);
            int gn = n0 + nt * 16 + r;
            if (gm < M && gn < N) {
                float v = acc[nt][j];
                if (flags & FLAG_BIAS)  v += bias[gn];
                if (flags & FLAG_RELU)  v = fmaxf(v, 0.f);
                if (flags & FLAG_AI)    v = ((gm == gn) ? alpha : 0.f) - v;
                if (flags & FLAG_SCALE) v *= alpha;
                if (flags & FLAG_ADD) {
                    int gr = gm - row_off;
                    if (gr >= 0) C[(size_t)gr * ldc + gn] += v;
                } else {
                    C[(size_t)gm * ldc + gn] = v;
                }
            }
        }
}

// ---------------------------------------------------------------- fused  a1 = softmax(q@kl^T); out = a1 @ W2
// 128 threads (4 waves) per (head, 64-token tile); each wave owns a 16-token strip.
// LDS (64KB) reused across phases:
//   phase 1: As = q tile [64][64] bf16 (8KB)   Kls = kl [256][64] bf16 (32KB)
//   phase 2: Pls = probs [64][256] bf16 (32KB) W2t = W2^T [64][256] bf16 (32KB)
__global__ void __launch_bounds__(128)
fused_a1_out(const float* __restrict__ qkv, const float* __restrict__ kl,
             const float* __restrict__ W2m, float* __restrict__ aout)
{
    __shared__ unsigned short smem[32768];   // 64 KB
    unsigned short* As  = smem;              // [64][64]
    unsigned short* Kls = smem + 4096;       // [256][64]
    unsigned short* Pls = smem;              // [64][256]
    unsigned short* W2t = smem + 16384;      // [64][256]

    const int tid = threadIdx.x, wave = tid >> 5, lane = tid & 31;
    const int r = lane & 15, half = lane >> 4;
    const int t0 = blockIdx.x * 64, head = blockIdx.y;

    // ---- stage q tile (scaled by DH^-0.5) : 64 tokens x 64 dims
    {
        int row = tid >> 1, seg = (tid & 1) * 32;
        const float4* src = (const float4*)(qkv + (size_t)(t0 + row) * QKVW + head * DH + seg);
        unsigned int* dst = (unsigned int*)(As + row * 64 + seg);
#pragma unroll
        for (int u = 0; u < 8; ++u) {
            float4 f = src[u];
            dst[2 * u]     = pk2(f.x * 0.125f, f.y * 0.125f);
            dst[2 * u + 1] = pk2(f.z * 0.125f, f.w * 0.125f);
        }
    }
    // ---- stage kl (256 x 64)
    {
#pragma unroll
        for (int rr = 0; rr < 2; ++rr) {
            int n = tid * 2 + rr;
            const float4* src = (const float4*)(kl + ((size_t)head * LM + n) * DH);
            unsigned int* dst = (unsigned int*)(Kls + n * 64);
#pragma unroll
            for (int u = 0; u < 16; ++u) {
                float4 f = src[u];
                dst[2 * u]     = pk2(f.x, f.y);
                dst[2 * u + 1] = pk2(f.z, f.w);
            }
        }
    }
    __syncthreads();

    // ---- stage 1: scores = q @ kl^T : (16 x 256 per wave)
    f32x8 acc[16];
    for (int t = 0; t < 16; ++t)
        for (int e = 0; e < 8; ++e) acc[t][e] = 0.f;

    for (int ks = 0; ks < 2; ++ks) {
        union { bf16x16 v; unsigned short u[16]; } af;
#pragma unroll
        for (int i = 0; i < 16; ++i) {
            int kk = ks * 32 + (half << 4) + i;
            af.u[i] = As[(wave * 16 + r) * 64 + kk];
        }
#pragma unroll
        for (int nt = 0; nt < 16; ++nt) {
            union { bf16x16 v; unsigned short u[16]; } bfr;
#pragma unroll
            for (int i = 0; i < 16; ++i) {
                int kk = ks * 32 + (half << 4) + i;
                bfr.u[i] = Kls[(nt * 16 + r) * 64 + kk];
            }
            acc[nt] = __builtin_amdgcn_wmma_f32_16x16x32_bf16(
                false, af.v, false, bfr.v, (short)0, acc[nt], false, false);
        }
    }

    // ---- row softmax over 256 (row M = j + 8*half; N spread over 16 lanes of the half)
#pragma unroll
    for (int j = 0; j < 8; ++j) {
        float m = acc[0][j];
#pragma unroll
        for (int t = 1; t < 16; ++t) m = fmaxf(m, acc[t][j]);
        for (int msk = 1; msk < 16; msk <<= 1) m = fmaxf(m, __shfl_xor(m, msk));
        float s = 0.f;
#pragma unroll
        for (int t = 0; t < 16; ++t) { float e = expf(acc[t][j] - m); acc[t][j] = e; s += e; }
        for (int msk = 1; msk < 16; msk <<= 1) s += __shfl_xor(s, msk);
        float inv = 1.f / s;
#pragma unroll
        for (int t = 0; t < 16; ++t) acc[t][j] *= inv;
    }

    __syncthreads();   // everyone done reading As/Kls before LDS reuse

    // ---- write probabilities (bf16, row-major [64][256])
#pragma unroll
    for (int nt = 0; nt < 16; ++nt)
#pragma unroll
        for (int j = 0; j < 8; ++j)
            Pls[(wave * 16 + j + 8 * half) * 256 + nt * 16 + r] = f2bf(acc[nt][j]);

    // ---- stage W2^T : W2t[n][k] = W2m[head][k][n]
    {
#pragma unroll
        for (int u = 0; u < 8; ++u) {
            int id = tid * 8 + u;              // 0..1023
            int k = id >> 2, nseg = (id & 3) * 16;
            const float4* src = (const float4*)(W2m + ((size_t)head * LM + k) * DH + nseg);
            float4 f0 = src[0], f1 = src[1], f2 = src[2], f3 = src[3];
            float v[16] = { f0.x,f0.y,f0.z,f0.w, f1.x,f1.y,f1.z,f1.w,
                            f2.x,f2.y,f2.z,f2.w, f3.x,f3.y,f3.z,f3.w };
#pragma unroll
            for (int j = 0; j < 16; ++j) W2t[(nseg + j) * 256 + k] = f2bf(v[j]);
        }
    }
    __syncthreads();

    // ---- stage 2: out = P(16x256) @ W2(256x64) per wave
    f32x8 oacc[4];
    for (int t = 0; t < 4; ++t)
        for (int e = 0; e < 8; ++e) oacc[t][e] = 0.f;

    for (int ks = 0; ks < 8; ++ks) {
        union { bf16x16 v; unsigned short u[16]; } af;
#pragma unroll
        for (int i = 0; i < 16; ++i) {
            int kk = ks * 32 + (half << 4) + i;
            af.u[i] = Pls[(wave * 16 + r) * 256 + kk];
        }
#pragma unroll
        for (int nt = 0; nt < 4; ++nt) {
            union { bf16x16 v; unsigned short u[16]; } bfr;
#pragma unroll
            for (int i = 0; i < 16; ++i) {
                int kk = ks * 32 + (half << 4) + i;
                bfr.u[i] = W2t[(nt * 16 + r) * 256 + kk];
            }
            oacc[nt] = __builtin_amdgcn_wmma_f32_16x16x32_bf16(
                false, af.v, false, bfr.v, (short)0, oacc[nt], false, false);
        }
    }
#pragma unroll
    for (int nt = 0; nt < 4; ++nt)
#pragma unroll
        for (int j = 0; j < 8; ++j) {
            int row = t0 + wave * 16 + j + 8 * half;
            aout[(size_t)row * DIMC + head * DH + nt * 16 + r] = oacc[nt][j];
        }
}

// ---------------------------------------------------------------- small kernels
__global__ void zero_f32(float* p, int n) {
    int i = blockIdx.x * 256 + threadIdx.x;
    if (i < n) p[i] = 0.f;
}

__global__ void fixup_h(float* h, const float* cls) {
    int idx = blockIdx.x * 256 + threadIdx.x;
    if (idx < 512) { h[idx] = cls[idx]; return; }
    idx -= 512;
    if (idx >= 129 * 512) return;
    int row = idx >> 9, c = idx & 511;
    h[(size_t)(N0 + 1 + row) * DIMC + c] = h[(size_t)(1 + row) * DIMC + c];
}

__global__ void layernorm_rows(const float* __restrict__ x, const float* g, const float* b,
                               float* __restrict__ y, int row_out_off)
{
    __shared__ float red[256];
    int row = blockIdx.x, tid = threadIdx.x;
    const float* xr = x + (size_t)row * DIMC;
    float a0 = xr[tid], a1 = xr[tid + 256];
    red[tid] = a0 + a1; __syncthreads();
    for (int s = 128; s > 0; s >>= 1) { if (tid < s) red[tid] += red[tid + s]; __syncthreads(); }
    float mu = red[0] * (1.f / DIMC); __syncthreads();
    float d0 = a0 - mu, d1 = a1 - mu;
    red[tid] = d0 * d0 + d1 * d1; __syncthreads();
    for (int s = 128; s > 0; s >>= 1) { if (tid < s) red[tid] += red[tid + s]; __syncthreads(); }
    float rs = rsqrtf(red[0] * (1.f / DIMC) + 1e-5f);
    float* yr = y + (size_t)(row + row_out_off) * DIMC;
    yr[tid]       = d0 * rs * g[tid]       + b[tid];
    yr[tid + 256] = d1 * rs * g[tid + 256] + b[tid + 256];
}

__global__ void landmarks(const float* __restrict__ qkv, float* __restrict__ ql, float* __restrict__ kl) {
    int idx = blockIdx.x * 256 + threadIdx.x;
    if (idx >= HEADS * LM * DH) return;
    int d = idx & 63, i = (idx >> 6) & 255, h = idx >> 14;
    float sq = 0.f, sk = 0.f;
    for (int t = 0; t < LPER; ++t) {
        size_t base = (size_t)(i * LPER + t) * QKVW + h * DH + d;
        sq += qkv[base];
        sk += qkv[base + DIMC];
    }
    ql[idx] = sq * 0.125f / LPER;
    kl[idx] = sk / LPER;
}

__global__ void softmax256(float* x) {
    __shared__ float red[256];
    int row = blockIdx.x, tid = threadIdx.x;
    float v = x[(size_t)row * 256 + tid];
    red[tid] = v; __syncthreads();
    for (int s = 128; s > 0; s >>= 1) { if (tid < s) red[tid] = fmaxf(red[tid], red[tid + s]); __syncthreads(); }
    float m = red[0]; __syncthreads();
    float e = expf(v - m);
    red[tid] = e; __syncthreads();
    for (int s = 128; s > 0; s >>= 1) { if (tid < s) red[tid] += red[tid + s]; __syncthreads(); }
    x[(size_t)row * 256 + tid] = e / red[0];
}

__global__ void abs_sums(const float* __restrict__ a2, float* cs, float* rs) {
    int idx = blockIdx.x * 256 + threadIdx.x;   // 2048
    int h = idx >> 8, i = idx & 255;
    const float* Ah = a2 + (size_t)h * 65536;
    float c = 0.f, r = 0.f;
    for (int j = 0; j < 256; ++j) { c += fabsf(Ah[i * 256 + j]); r += fabsf(Ah[j * 256 + i]); }
    cs[idx] = c; rs[idx] = r;
}

__global__ void max_scale(const float* cs, const float* rs, float* inv) {
    __shared__ float r1[256], r2[256];
    int tid = threadIdx.x;
    float m1 = 0.f, m2 = 0.f;
    for (int i = tid; i < 2048; i += 256) { m1 = fmaxf(m1, cs[i]); m2 = fmaxf(m2, rs[i]); }
    r1[tid] = m1; r2[tid] = m2; __syncthreads();
    for (int s = 128; s > 0; s >>= 1) {
        if (tid < s) { r1[tid] = fmaxf(r1[tid], r1[tid + s]); r2[tid] = fmaxf(r2[tid], r2[tid + s]); }
        __syncthreads();
    }
    if (tid == 0) inv[0] = 1.f / (r1[0] * r2[0]);
}

__global__ void build_z0(const float* __restrict__ a2, const float* inv, float* __restrict__ z) {
    int idx = blockIdx.x * 256 + threadIdx.x;
    if (idx >= HEADS * 65536) return;
    int h = idx >> 16, rem = idx & 65535, i = rem >> 8, j = rem & 255;
    z[idx] = a2[((size_t)h << 16) + (j << 8) + i] * inv[0];
}

__global__ void ew_aI_minus(const float* __restrict__ x, float a, float* __restrict__ y) {
    int idx = blockIdx.x * 256 + threadIdx.x;
    if (idx >= HEADS * 65536) return;
    int rem = idx & 65535, i = rem >> 8, j = rem & 255;
    y[idx] = ((i == j) ? a : 0.f) - x[idx];
}

// a3v[h][i][:] = softmax(ql[h][i] @ k^T) @ v    (streamed over 15360 tokens, float4 dots)
__global__ void __launch_bounds__(256)
a3v_kernel(const float* __restrict__ ql, const float* __restrict__ qkv, float* __restrict__ a3v) {
    __shared__ float qlrow[DH];
    __shared__ float red[256];
    __shared__ float wred[8];
    int i = blockIdx.x, h = blockIdx.y, tid = threadIdx.x;
    if (tid < DH) qlrow[tid] = ql[(size_t)(h * LM + i) * DH + tid];
    __syncthreads();

    float lmax = -3.4e38f;
    for (int t = tid; t < NP; t += 256) {
        const float4* kr = (const float4*)(qkv + (size_t)t * QKVW + DIMC + h * DH);
        float s = 0.f;
#pragma unroll
        for (int d = 0; d < 16; ++d) {
            float4 kv = kr[d];
            s += qlrow[4 * d] * kv.x + qlrow[4 * d + 1] * kv.y
               + qlrow[4 * d + 2] * kv.z + qlrow[4 * d + 3] * kv.w;
        }
        lmax = fmaxf(lmax, s);
    }
    red[tid] = lmax; __syncthreads();
    for (int s = 128; s > 0; s >>= 1) { if (tid < s) red[tid] = fmaxf(red[tid], red[tid + s]); __syncthreads(); }
    float gmax = red[0]; __syncthreads();

    float acc[DH];
    for (int d = 0; d < DH; ++d) acc[d] = 0.f;
    float lsum = 0.f;
    for (int t = tid; t < NP; t += 256) {
        const float4* kr = (const float4*)(qkv + (size_t)t * QKVW + DIMC + h * DH);
        float s = 0.f;
#pragma unroll
        for (int d = 0; d < 16; ++d) {
            float4 kv = kr[d];
            s += qlrow[4 * d] * kv.x + qlrow[4 * d + 1] * kv.y
               + qlrow[4 * d + 2] * kv.z + qlrow[4 * d + 3] * kv.w;
        }
        float e = expf(s - gmax);
        lsum += e;
        const float4* vr = (const float4*)(qkv + (size_t)t * QKVW + 2 * DIMC + h * DH);
#pragma unroll
        for (int d = 0; d < 16; ++d) {
            float4 vv = vr[d];
            acc[4 * d] += e * vv.x; acc[4 * d + 1] += e * vv.y;
            acc[4 * d + 2] += e * vv.z; acc[4 * d + 3] += e * vv.w;
        }
    }
    red[tid] = lsum; __syncthreads();
    for (int s = 128; s > 0; s >>= 1) { if (tid < s) red[tid] += red[tid + s]; __syncthreads(); }
    float ginv = 1.f / red[0]; __syncthreads();

    int lane = tid & 31, wv = tid >> 5;
    for (int d = 0; d < DH; ++d) {
        float a = acc[d];
        for (int m = 16; m >= 1; m >>= 1) a += __shfl_xor(a, m);
        if (lane == 0) wred[wv] = a;
        __syncthreads();
        if (tid == 0) {
            float s = 0.f;
            for (int w = 0; w < 8; ++w) s += wred[w];
            a3v[(size_t)(h * LM + i) * DH + d] = s * ginv;
        }
        __syncthreads();
    }
}

__global__ void res_conv_add(const float* __restrict__ qkv, const float* __restrict__ rw,
                             float* __restrict__ aout) {
    int idx = blockIdx.x * 256 + threadIdx.x;
    if (idx >= NP * DIMC) return;
    int c = idx & 511, t = idx >> 9;
    int h = c >> 6, d = c & 63;
    float s = 0.f;
    for (int kk = 0; kk < RESK; ++kk) {
        int tt = t + kk - RESK / 2;
        if (tt >= 0 && tt < NP)
            s += qkv[(size_t)tt * QKVW + 2 * DIMC + h * DH + d] * rw[h * RESK + kk];
    }
    aout[idx] += s;
}

__global__ void ppeg_kernel(const float* __restrict__ h,
                            const float* __restrict__ w7, const float* __restrict__ b7,
                            const float* __restrict__ w5, const float* __restrict__ b5,
                            const float* __restrict__ w3, const float* __restrict__ b3,
                            float* __restrict__ hp) {
    int idx = blockIdx.x * 256 + threadIdx.x;
    if (idx >= NFEAT * DIMC) return;
    int c = idx & 511, p = idx >> 9;
    int r = p / HW_, cc = p % HW_;
    float s = h[(size_t)(1 + p) * DIMC + c] + b7[c] + b5[c] + b3[c];
    const float* W7 = w7 + c * 49;
    for (int ky = 0; ky < 7; ++ky)
        for (int kx = 0; kx < 7; ++kx) {
            int rr = r + ky - 3, c2 = cc + kx - 3;
            if (rr >= 0 && rr < HW_ && c2 >= 0 && c2 < HW_)
                s += W7[ky * 7 + kx] * h[(size_t)(1 + rr * HW_ + c2) * DIMC + c];
        }
    const float* W5 = w5 + c * 25;
    for (int ky = 0; ky < 5; ++ky)
        for (int kx = 0; kx < 5; ++kx) {
            int rr = r + ky - 2, c2 = cc + kx - 2;
            if (rr >= 0 && rr < HW_ && c2 >= 0 && c2 < HW_)
                s += W5[ky * 5 + kx] * h[(size_t)(1 + rr * HW_ + c2) * DIMC + c];
        }
    const float* W3 = w3 + c * 9;
    for (int ky = 0; ky < 3; ++ky)
        for (int kx = 0; kx < 3; ++kx) {
            int rr = r + ky - 1, c2 = cc + kx - 1;
            if (rr >= 0 && rr < HW_ && c2 >= 0 && c2 < HW_)
                s += W3[ky * 3 + kx] * h[(size_t)(1 + rr * HW_ + c2) * DIMC + c];
        }
    hp[(size_t)(1 + p) * DIMC + c] = s;
}

__global__ void copy512(const float* src, float* dst) {
    int i = blockIdx.x * 256 + threadIdx.x;
    if (i < 512) dst[i] = src[i];
}

__global__ void final_head(const float* __restrict__ h, const float* g, const float* b,
                           const float* __restrict__ W, const float* __restrict__ bias,
                           float* __restrict__ out) {
    __shared__ float red[256];
    __shared__ float lg[5];
    int tid = threadIdx.x;
    float a0 = h[tid], a1 = h[tid + 256];
    red[tid] = a0 + a1; __syncthreads();
    for (int s = 128; s > 0; s >>= 1) { if (tid < s) red[tid] += red[tid + s]; __syncthreads(); }
    float mu = red[0] * (1.f / DIMC); __syncthreads();
    float d0 = a0 - mu, d1 = a1 - mu;
    red[tid] = d0 * d0 + d1 * d1; __syncthreads();
    for (int s = 128; s > 0; s >>= 1) { if (tid < s) red[tid] += red[tid + s]; __syncthreads(); }
    float rs = rsqrtf(red[0] * (1.f / DIMC) + 1e-5f); __syncthreads();
    float n0 = d0 * rs * g[tid] + b[tid];
    float n1 = d1 * rs * g[tid + 256] + b[tid + 256];
    for (int j = 0; j < 5; ++j) {
        red[tid] = n0 * W[tid * 5 + j] + n1 * W[(tid + 256) * 5 + j];
        __syncthreads();
        for (int s = 128; s > 0; s >>= 1) { if (tid < s) red[tid] += red[tid + s]; __syncthreads(); }
        if (tid == 0) lg[j] = red[0] + bias[j];
        __syncthreads();
    }
    if (tid == 0) {
        float mx = lg[0]; int am = 0;
        for (int j = 1; j < 5; ++j) if (lg[j] > mx) { mx = lg[j]; am = j; }
        float e[5], se = 0.f;
        for (int j = 0; j < 5; ++j) { e[j] = expf(lg[j] - mx); se += e[j]; }
        for (int j = 0; j < 5; ++j) out[j] = lg[j];
        for (int j = 0; j < 5; ++j) out[5 + j] = e[j] / se;
        out[10] = (float)am;
    }
}

// ---------------------------------------------------------------- host side
static void gemm(hipStream_t s, const float* A, int lda, long long sA,
                 const float* B, int ldbk, int ldbn, long long sB,
                 float* C, int ldc, long long sC, const float* bias,
                 int M, int N, int K, int flags, float alpha, int row_off, int batch) {
    dim3 g((N + 63) / 64, (M + 63) / 64, batch);
    gemm_wmma<<<g, 128, 0, s>>>(A, lda, sA, B, ldbk, ldbn, sB, C, ldc, sC,
                                bias, M, N, K, flags, alpha, row_off);
}

struct Bufs {
    float *h, *hp, *xln, *qkv, *aout, *ql, *kl, *sim2;
    float *cs, *rs, *inv, *zA, *zB, *xz, *t1, *t3, *a3v, *W2m;
};

static void run_layer(hipStream_t s, float* hbuf,
                      const float* lng, const float* lnb,
                      const float* Wqkv, const float* Wout, const float* bout,
                      const float* resw, const Bufs& B) {
    zero_f32<<<(PADR * DIMC + 255) / 256, 256, 0, s>>>(B.xln, PADR * DIMC);
    layernorm_rows<<<NTOK, 256, 0, s>>>(hbuf, lng, lnb, B.xln, PADR);

    gemm(s, B.xln, DIMC, 0, Wqkv, QKVW, 1, 0, B.qkv, QKVW, 0, nullptr,
         NP, QKVW, DIMC, 0, 0.f, 0, 1);

    landmarks<<<(HEADS * LM * DH + 255) / 256, 256, 0, s>>>(B.qkv, B.ql, B.kl);

    // sim2 = ql @ kl^T   (per head; B contiguous along k -> ldbk==1 path)
    gemm(s, B.ql, DH, LM * DH, B.kl, 1, DH, LM * DH, B.sim2, LM, LM * LM, nullptr,
         LM, LM, DH, 0, 0.f, 0, HEADS);
    softmax256<<<HEADS * LM, 256, 0, s>>>(B.sim2);

    // Moore-Penrose pinv
    abs_sums<<<8, 256, 0, s>>>(B.sim2, B.cs, B.rs);
    max_scale<<<1, 256, 0, s>>>(B.cs, B.rs, B.inv);
    build_z0<<<(HEADS * LM * LM + 255) / 256, 256, 0, s>>>(B.sim2, B.inv, B.zA);
    float* zc = B.zA; float* zn = B.zB;
    for (int it = 0; it < 6; ++it) {
        gemm(s, B.sim2, LM, LM * LM, zc, LM, 1, LM * LM, B.xz, LM, LM * LM, nullptr,
             LM, LM, LM, 0, 0.f, 0, HEADS);
        ew_aI_minus<<<(HEADS * LM * LM + 255) / 256, 256, 0, s>>>(B.xz, 7.f, B.t1);
        gemm(s, B.xz, LM, LM * LM, B.t1, LM, 1, LM * LM, B.t3, LM, LM * LM, nullptr,
             LM, LM, LM, FLAG_AI, 15.f, 0, HEADS);
        gemm(s, B.xz, LM, LM * LM, B.t3, LM, 1, LM * LM, B.t1, LM, LM * LM, nullptr,
             LM, LM, LM, FLAG_AI, 13.f, 0, HEADS);   // t5 in t1
        gemm(s, zc, LM, LM * LM, B.t1, LM, 1, LM * LM, zn, LM, LM * LM, nullptr,
             LM, LM, LM, FLAG_SCALE, 0.25f, 0, HEADS);
        float* tmp = zc; zc = zn; zn = tmp;
    }

    a3v_kernel<<<dim3(LM, HEADS), 256, 0, s>>>(B.ql, B.qkv, B.a3v);

    // W2 = a2_inv @ (a3 @ v)
    gemm(s, zc, LM, LM * LM, B.a3v, DH, 1, LM * DH, B.W2m, DH, LM * DH, nullptr,
         LM, DH, LM, 0, 0.f, 0, HEADS);

    fused_a1_out<<<dim3(NP / 64, HEADS), 128, 0, s>>>(B.qkv, B.kl, B.W2m, B.aout);
    res_conv_add<<<(NP * DIMC + 255) / 256, 256, 0, s>>>(B.qkv, resw, B.aout);

    // h += (attn_out @ Wout + bout)[-NTOK:]
    gemm(s, B.aout, DIMC, 0, Wout, DIMC, 1, 0, hbuf, DIMC, 0, bout,
         NP, DIMC, DIMC, FLAG_BIAS | FLAG_ADD, 0.f, PADR, 1);
}

extern "C" void kernel_launch(void* const* d_in, const int* in_sizes, int n_in,
                              void* d_out, int out_size, void* d_ws, size_t ws_size,
                              hipStream_t stream) {
    const float* data    = (const float*)d_in[0];
    const float* W_fc1   = (const float*)d_in[1];
    const float* b_fc1   = (const float*)d_in[2];
    const float* cls_tok = (const float*)d_in[3];
    const float* ln1_g   = (const float*)d_in[4];
    const float* ln1_b   = (const float*)d_in[5];
    const float* qkv1    = (const float*)d_in[6];
    const float* out1_w  = (const float*)d_in[7];
    const float* out1_b  = (const float*)d_in[8];
    const float* res1    = (const float*)d_in[9];
    const float* w7      = (const float*)d_in[10];
    const float* b7      = (const float*)d_in[11];
    const float* w5      = (const float*)d_in[12];
    const float* b5      = (const float*)d_in[13];
    const float* w3      = (const float*)d_in[14];
    const float* b3      = (const float*)d_in[15];
    const float* ln2_g   = (const float*)d_in[16];
    const float* ln2_b   = (const float*)d_in[17];
    const float* qkv2    = (const float*)d_in[18];
    const float* out2_w  = (const float*)d_in[19];
    const float* out2_b  = (const float*)d_in[20];
    const float* res2    = (const float*)d_in[21];
    const float* norm_g  = (const float*)d_in[22];
    const float* norm_b  = (const float*)d_in[23];
    const float* W_fc2   = (const float*)d_in[24];
    const float* b_fc2   = (const float*)d_in[25];

    float* ws = (float*)d_ws;
    Bufs B;
    size_t o = 0;
    B.h    = ws + o; o += (size_t)NTOK * DIMC;
    B.hp   = ws + o; o += (size_t)NTOK * DIMC;
    B.xln  = ws + o; o += (size_t)NP * DIMC;
    B.qkv  = ws + o; o += (size_t)NP * QKVW;
    B.aout = ws + o; o += (size_t)NP * DIMC;
    B.ql   = ws + o; o += (size_t)HEADS * LM * DH;
    B.kl   = ws + o; o += (size_t)HEADS * LM * DH;
    B.sim2 = ws + o; o += (size_t)HEADS * LM * LM;
    B.cs   = ws + o; o += 2048;
    B.rs   = ws + o; o += 2048;
    B.inv  = ws + o; o += 16;
    B.zA   = ws + o; o += (size_t)HEADS * LM * LM;
    B.zB   = ws + o; o += (size_t)HEADS * LM * LM;
    B.xz   = ws + o; o += (size_t)HEADS * LM * LM;
    B.t1   = ws + o; o += (size_t)HEADS * LM * LM;
    B.t3   = ws + o; o += (size_t)HEADS * LM * LM;
    B.a3v  = ws + o; o += (size_t)HEADS * LM * DH;
    B.W2m  = ws + o; o += (size_t)HEADS * LM * DH;

    // h[1:15001] = relu(data @ W_fc1 + b), then cls + wraparound rows
    gemm(stream, data, 1024, 0, W_fc1, DIMC, 1, 0, B.h + DIMC, DIMC, 0, b_fc1,
         N0, DIMC, 1024, FLAG_BIAS | FLAG_RELU, 0.f, 0, 1);
    fixup_h<<<(130 * 512 + 255) / 256, 256, 0, stream>>>(B.h, cls_tok);

    run_layer(stream, B.h, ln1_g, ln1_b, qkv1, out1_w, out1_b, res1, B);

    ppeg_kernel<<<((size_t)NFEAT * DIMC + 255) / 256, 256, 0, stream>>>(
        B.h, w7, b7, w5, b5, w3, b3, B.hp);
    copy512<<<2, 256, 0, stream>>>(B.h, B.hp);

    run_layer(stream, B.hp, ln2_g, ln2_b, qkv2, out2_w, out2_b, res2, B);

    final_head<<<1, 256, 0, stream>>>(B.hp, norm_g, norm_b, W_fc2, b_fc2, (float*)d_out);
}
